// Actor_87359634801144
// MI455X (gfx1250) — matrix operand order repaired and checked by reading, not compiled, not run on hardware
//
#include <hip/hip_runtime.h>

#define STATE_DIM 512
#define AFEAT 64
#define HID 256
#define HID2 128
#define BATCH 256
#define KACT 1000
#define ROWS_BLK 128           // rows (b,k) per block
#define KTILES 8               // ceil(1000/128)
#define HSB_BTILE 16           // batch rows per hsb block

typedef __attribute__((ext_vector_type(16))) __bf16 v16bf;
typedef __attribute__((ext_vector_type(8)))  __bf16 v8bf;
typedef __attribute__((ext_vector_type(4)))  __bf16 v4bf;
typedef __attribute__((ext_vector_type(8)))  float  v8f;

// workspace layout (bytes)
#define HSB_OFF  0                          // 256*256 f32 = 262144 B
#define W1AF_OFF (BATCH * HID * 4)          // 16384 bf16 A-frags (W1_a^T)
#define W2F_OFF  (W1AF_OFF + 16384 * 2)     // 32768 bf16 A-frags (W2^T)
#define W3_OFF   (W2F_OFF + 32768 * 2)      // 128 bf16

// single-instruction ReLU: v_med3_f32(x, 0, +inf); WMMA never emits sNaN
__device__ __forceinline__ float relu1(float x) {
    return __builtin_amdgcn_fmed3f(x, 0.f, __builtin_inff());
}

// ---------------------------------------------------------------------------
// hsb[b][n] = state_embed[b,:] . W1[:512, n] + b1[n]
// Tiled: 16 blocks, each covers 16 batch rows x all 256 columns. W1_s is
// streamed once per block (16x less L2 traffic than one-block-per-b).
__global__ __launch_bounds__(256) void hsb_kernel(
        const float* __restrict__ se,
        const float* __restrict__ W1,
        const float* __restrict__ b1,
        float* __restrict__ hsb) {
    __shared__ float ses[HSB_BTILE][STATE_DIM];     // 32 KB
    const int n  = threadIdx.x;                     // output column 0..255
    const int b0 = blockIdx.x * HSB_BTILE;

    for (int idx = threadIdx.x; idx < HSB_BTILE * STATE_DIM; idx += 256) {
        const int bl = idx >> 9;
        const int i  = idx & (STATE_DIM - 1);
        ses[bl][i] = se[(size_t)(b0 + bl) * STATE_DIM + i];
    }
    __syncthreads();

    float acc[HSB_BTILE];
    const float bias = b1[n];
#pragma unroll
    for (int bl = 0; bl < HSB_BTILE; ++bl) acc[bl] = bias;

#pragma unroll 4
    for (int i = 0; i < STATE_DIM; ++i) {
        const float wv = W1[(size_t)i * HID + n];   // coalesced; LDS broadcast below
#pragma unroll
        for (int bl = 0; bl < HSB_BTILE; ++bl)
            acc[bl] += ses[bl][i] * wv;
    }
#pragma unroll
    for (int bl = 0; bl < HSB_BTILE; ++bl)
        hsb[(size_t)(b0 + bl) * HID + n] = acc[bl];
}

// ---------------------------------------------------------------------------
// Pre-swizzle weights into WMMA *A*-fragment order (bf16), A = W^T tiles.
// A layout (16x32 bf16): lane l holds matrix row (=output col n) l&15;
// packed element e maps to K = hk+e, hk=(l<16?0:16) (this code's contiguous k order; B loads match).
__device__ __forceinline__ int kmap(int e, int lane) {
    const int hk = (lane < 16) ? 0 : 16;
    return hk + e;
}

__global__ void prep_kernel(const float* __restrict__ W1,
                            const float* __restrict__ W2,
                            const float* __restrict__ W3,
                            __bf16* __restrict__ w1af,
                            __bf16* __restrict__ w2f,
                            __bf16* __restrict__ w3b) {
    const int t = blockIdx.x * blockDim.x + threadIdx.x;
    if (t < 16384) {                 // W1_a^T: 16 n-tiles x 2 k-chunks
        const int e    = t & 15;
        const int lane = (t >> 4) & 31;
        const int kc   = (t >> 9) & 1;
        const int nt   = (t >> 10) & 15;
        const int K = kc * 32 + kmap(e, lane);      // feature index (0..63)
        const int N = nt * 16 + (lane & 15);        // hidden index
        w1af[t] = (__bf16)W1[(size_t)(STATE_DIM + K) * HID + N];
    } else if (t < 16384 + 32768) {  // W2^T: 8 n-tiles x 8 k-chunks
        const int u    = t - 16384;
        const int e    = u & 15;
        const int lane = (u >> 4) & 31;
        const int kc   = (u >> 9) & 7;
        const int n2   = (u >> 12) & 7;
        const int K = kc * 32 + kmap(e, lane);      // hidden index (0..255)
        const int N = n2 * 16 + (lane & 15);        // hidden2 index
        w2f[u] = (__bf16)W2[(size_t)K * HID2 + N];
    } else if (t < 16384 + 32768 + HID2) {
        const int u = t - 16384 - 32768;
        w3b[u] = (__bf16)W3[u];
    }
}

// ---------------------------------------------------------------------------
// Fused MLP, weight-stationary: D = (W^T)(X^T) so weights are the A matrix
// (registers, loaded once per block) and activations stream through LDS as
// contiguous B fragments. 8 waves; wave w owns n-tiles {2w,2w+1} (layer 1)
// and n-tile w (layer 2), sweeping all 8 row-tiles of the block.
__global__ __launch_bounds__(256) void actor_kernel(
        const float* __restrict__ feats,
        const float* __restrict__ b2,
        const float* __restrict__ b3,
        const float* __restrict__ hsb,
        const __bf16* __restrict__ w1af,
        const __bf16* __restrict__ w2f,
        const __bf16* __restrict__ w3b,
        float* __restrict__ out) {
    __shared__ __bf16 H1s[ROWS_BLK * HID];          // 64 KB, row-major [row][h]
    __shared__ __bf16 uni[ROWS_BLK * HID2];         // 32 KB: featsT then H2
    __shared__ float  W3s[HID2];

    const int tid  = threadIdx.x;
    const int w    = tid >> 5;                      // wave id 0..7 (wave32)
    const int lane = tid & 31;
    const int m    = lane & 15;
    const int hk   = (lane < 16) ? 0 : 8;
    const int ofs  = (lane < 16) ? 0 : 16;
    const int b    = blockIdx.y;
    const int kb   = blockIdx.x * ROWS_BLK;

    // ---- stage action features (f32 -> bf16) into LDS, zero-fill tail rows
#pragma unroll
    for (int it = 0; it < 8; ++it) {
        const int idx4 = tid + it * 256;            // 2048 float4 tiles
        const int r    = idx4 >> 4;                 // row 0..127
        const int f4   = (idx4 & 15) << 2;          // feature 0..60 step 4
        const int kr   = kb + r;
        float4 v = make_float4(0.f, 0.f, 0.f, 0.f);
        if (kr < KACT)
            v = *(const float4*)(feats + ((size_t)b * KACT + kr) * AFEAT + f4);
        v4bf h;
        h[0] = (__bf16)v.x; h[1] = (__bf16)v.y;
        h[2] = (__bf16)v.z; h[3] = (__bf16)v.w;
        *(v4bf*)&uni[r * AFEAT + f4] = h;
    }
    if (tid < HID2) W3s[tid] = (float)w3b[tid];

    // ---- weight A-fragments: loaded exactly once per block across waves
    v16bf wf1[2][2];
#pragma unroll
    for (int nt2 = 0; nt2 < 2; ++nt2)
#pragma unroll
        for (int kc = 0; kc < 2; ++kc)
            wf1[nt2][kc] = *(const v16bf*)(
                w1af + ((((2 * w + nt2) * 2) + kc) * 32 + lane) * 16);
    v16bf wf2[8];
#pragma unroll
    for (int kc = 0; kc < 8; ++kc)
        wf2[kc] = *(const v16bf*)(w2f + ((w * 8 + kc) * 32 + lane) * 16);

    // C inits: D row (vgpr i) = output index i+hk within the 16-wide n-tile
    v8f cinit1[2];
#pragma unroll
    for (int nt2 = 0; nt2 < 2; ++nt2)
#pragma unroll
        for (int i = 0; i < 8; ++i)
            cinit1[nt2][i] = hsb[(size_t)b * HID + (2 * w + nt2) * 16 + i + hk];
    v8f cinit2;
#pragma unroll
    for (int i = 0; i < 8; ++i) cinit2[i] = b2[w * 16 + i + hk];

    __syncthreads();

    // ---- Layer 1: D[n][m] = sum_k W1a^T[n][k] * feats[m][k]  (+hsb, ReLU)
    // Software-pipelined: next row-tile's B fragments load while the current
    // tile's WMMA->VALU hazard window drains.
    v16bf bf0 = *(const v16bf*)&uni[m * AFEAT + ofs];
    v16bf bf1 = *(const v16bf*)&uni[m * AFEAT + 32 + ofs];
#pragma unroll
    for (int rt = 0; rt < 8; ++rt) {
        const int rrow = rt * 16 + m;
        v16bf cur0 = bf0, cur1 = bf1;
        if (rt < 7) {
            bf0 = *(const v16bf*)&uni[(rrow + 16) * AFEAT + ofs];
            bf1 = *(const v16bf*)&uni[(rrow + 16) * AFEAT + 32 + ofs];
        }
#pragma unroll
        for (int nt2 = 0; nt2 < 2; ++nt2) {
            v8f c = cinit1[nt2];
            c = __builtin_amdgcn_wmma_f32_16x16x32_bf16(
                    false, wf1[nt2][0], false, cur0, (short)0, c, false, false);
            c = __builtin_amdgcn_wmma_f32_16x16x32_bf16(
                    false, wf1[nt2][1], false, cur1, (short)0, c, false, false);
            v8bf hv;
#pragma unroll
            for (int i = 0; i < 8; ++i)
                hv[i] = (__bf16)relu1(c[i]);
            // 8 contiguous h values -> one 16B LDS store
            *(v8bf*)&H1s[rrow * HID + (2 * w + nt2) * 16 + hk] = hv;
        }
    }
    __syncthreads();   // H1 complete; featsT region now dead

    // ---- Layer 2: D[n][m] = sum_k W2^T[n][k] * H1[m][k]  (+b2, ReLU)
#pragma unroll
    for (int rt = 0; rt < 8; ++rt) {
        const int rrow = rt * 16 + m;
        v8f c = cinit2;
#pragma unroll
        for (int kc = 0; kc < 8; ++kc) {
            v16bf bf = *(const v16bf*)&H1s[rrow * HID + kc * 32 + ofs];
            c = __builtin_amdgcn_wmma_f32_16x16x32_bf16(
                    false, wf2[kc], false, bf, (short)0, c, false, false);
        }
        v8bf hv;
#pragma unroll
        for (int i = 0; i < 8; ++i)
            hv[i] = (__bf16)relu1(c[i]);
        *(v8bf*)&uni[rrow * HID2 + w * 16 + hk] = hv;   // H2 overlays featsT
    }
    __syncthreads();

    // ---- Layer 3: logits = H2 . W3 + b3 (N=1 -> VALU), bank-swizzled
    if (tid < ROWS_BLK) {
        const int r = tid;
        float acc = b3[0];
#pragma unroll 8
        for (int j = 0; j < HID2; ++j) {
            const int cidx = (r * 8 + j) & (HID2 - 1);
            acc += (float)uni[r * HID2 + cidx] * W3s[cidx];
        }
        const int kr = kb + r;
        if (kr < KACT) out[(size_t)b * KACT + kr] = acc;
    }
}

// ---------------------------------------------------------------------------
extern "C" void kernel_launch(void* const* d_in, const int* in_sizes, int n_in,
                              void* d_out, int out_size, void* d_ws, size_t ws_size,
                              hipStream_t stream) {
    (void)in_sizes; (void)n_in; (void)out_size; (void)ws_size;
    const float* se    = (const float*)d_in[0];
    const float* feats = (const float*)d_in[1];
    const float* W1    = (const float*)d_in[2];
    const float* b1    = (const float*)d_in[3];
    const float* W2    = (const float*)d_in[4];
    const float* b2    = (const float*)d_in[5];
    const float* W3    = (const float*)d_in[6];
    const float* b3    = (const float*)d_in[7];
    float* out = (float*)d_out;

    char* ws = (char*)d_ws;
    float*  hsb  = (float*)(ws + HSB_OFF);
    __bf16* w1af = (__bf16*)(ws + W1AF_OFF);
    __bf16* w2f  = (__bf16*)(ws + W2F_OFF);
    __bf16* w3b  = (__bf16*)(ws + W3_OFF);

    hsb_kernel<<<dim3(BATCH / HSB_BTILE), dim3(256), 0, stream>>>(se, W1, b1, hsb);
    prep_kernel<<<dim3((16384 + 32768 + HID2 + 255) / 256), dim3(256), 0, stream>>>(
        W1, W2, W3, w1af, w2f, w3b);
    actor_kernel<<<dim3(KTILES, BATCH), dim3(256), 0, stream>>>(
        feats, b2, b3, hsb, w1af, w2f, w3b, out);
}
